// CRF_35476429865826
// MI455X (gfx1250) — matrix operand order repaired and checked by reading, not compiled, not run on hardware
//
#include <hip/hip_runtime.h>

// CRF NLL for MI455X (gfx1250, wave32).
// Kernel 1 (crf_score): coalesced scan of y/p/mask -> s1+s2 per batch into d_ws.
// Kernel 2 (crf_forward): scaled-forward recursion; q.E done with
//   v_wmma_f32_16x16x32_f16 (f16 in, f32 accumulate). 1 wave = 16 batches.
//   out[b] = sum_{t<L} log c_t - (s1+s2).

typedef __attribute__((ext_vector_type(16))) _Float16 v16h;
typedef __attribute__((ext_vector_type(8)))  float    v8f;

constexpr int kB = 256;
constexpr int kT = 2048;
constexpr int kD = 64;
constexpr int kMB = 16;     // batches per block (WMMA M)
constexpr int kROW = 68;    // padded LDS row stride (floats): 16B aligned, bank-rotated

// ---------------------------------------------------------------------------
// Kernel 2: scaled CRF forward with WMMA
// ---------------------------------------------------------------------------
__global__ __launch_bounds__(32)
void crf_forward(const float* __restrict__ p,
                 const int*   __restrict__ mask,
                 const float* __restrict__ transition,
                 const float* __restrict__ s_ws,
                 float*       __restrict__ out)
{
    __shared__ float uls[kMB * kROW];   // u_t, [16 batches][64 states] padded

    const int lane  = threadIdx.x;      // 0..31
    const int lo    = lane & 15;
    const int hi    = lane >> 4;        // half-wave id
    const int bbase = blockIdx.x * kMB;

    // ---- E = exp(transition) in WMMA B-operand layout (loop invariant) ----
    // B 32x16 f16 layout, matching A: element h<8 maps to K = hi*8 + h,
    // element h>=8 maps to K = 16 + hi*8 + (h-8) (+ 32*kchunk). N = nt*16 + lo.
    v16h breg[4][2];
    #pragma unroll
    for (int nt = 0; nt < 4; ++nt)
      #pragma unroll
      for (int kc = 0; kc < 2; ++kc) {
        const int n = nt * 16 + lo;
        #pragma unroll
        for (int h = 0; h < 16; ++h) {
          const int k = kc * 32 + (h < 8 ? hi * 8 + h : 16 + hi * 8 + (h - 8));   // d index (row of transition)
          breg[nt][kc][h] = (_Float16)__expf(transition[k * kD + n]);
        }
      }

    // ---- t = 0 : u_0 = exp(p[:,0,:]) written in C-layout order ----
    #pragma unroll
    for (int j = 0; j < 8; ++j) {
      const int row = hi * 8 + j;                // batch within block
      const float* prow = p + (size_t)(bbase + row) * (kT * kD);
      #pragma unroll
      for (int nt = 0; nt < 4; ++nt) {
        const int e = nt * 16 + lo;
        uls[row * kROW + e] = __expf(prow[e]);
      }
    }
    float vld  = 0.f;                            // valid[t-1] for lanes hi==0
    if (hi == 0) vld = 1.0f - (float)mask[(bbase + lo) * kT + 0];
    float zacc = 0.f;                            // sum of log c_t (lane lo = batch)

    for (int t = 1; t < kT; ++t) {
      asm volatile("" ::: "memory");             // order LDS phases (HW DS is in-order)

      // ---- consume u_{t-1}: row sums, normalize to q in WMMA A layout ----
      // A 16x32 f16 layout (ISA 7.12.2): lane<16 holds K = {0..7,16..23},
      // lane>=16 holds K = {8..15,24..31}; +32 for the second K-chunk.
      const float* rowp = &uls[lo * kROW];
      float g0[16], g1[16];
      float csum = 0.f;
      #pragma unroll
      for (int i = 0; i < 8; ++i) {
        g0[i]     = rowp[      hi * 8 + i];      // kc=0 group 0
        g0[8 + i] = rowp[ 16 + hi * 8 + i];      // kc=0 group 1
        g1[i]     = rowp[ 32 + hi * 8 + i];      // kc=1 group 0
        g1[8 + i] = rowp[ 48 + hi * 8 + i];      // kc=1 group 1
        csum += g0[i] + g0[8 + i] + g1[i] + g1[8 + i];
      }
      csum += __shfl_xor(csum, 16, 32);          // combine half-waves -> c_{t-1}[b]
      const float rinv = __builtin_amdgcn_rcpf(csum);
      if (hi == 0) zacc += vld * __logf(csum);

      v16h a0, a1;
      #pragma unroll
      for (int h = 0; h < 16; ++h) {
        a0[h] = (_Float16)(g0[h] * rinv);
        a1[h] = (_Float16)(g1[h] * rinv);
      }

      asm volatile("" ::: "memory");

      // ---- produce: S = q . E via 8 WMMAs; u_t = exp(p_t) * S ----
      v8f acc[4];
      #pragma unroll
      for (int nt = 0; nt < 4; ++nt) {
        v8f z = {0.f, 0.f, 0.f, 0.f, 0.f, 0.f, 0.f, 0.f};
        acc[nt] = __builtin_amdgcn_wmma_f32_16x16x32_f16(
            false, a0, false, breg[nt][0], (short)0, z, false, false);
        acc[nt] = __builtin_amdgcn_wmma_f32_16x16x32_f16(
            false, a1, false, breg[nt][1], (short)0, acc[nt], false, false);
      }

      #pragma unroll
      for (int j = 0; j < 8; ++j) {
        const int row = hi * 8 + j;              // C layout: VGPR j -> M = j + hi*8
        const float* prow = p + (size_t)(bbase + row) * (kT * kD) + (size_t)t * kD;
        #pragma unroll
        for (int nt = 0; nt < 4; ++nt) {
          const int e = nt * 16 + lo;            // C layout: lane -> N
          uls[row * kROW + e] = __expf(prow[e]) * acc[nt][j];
        }
      }

      if (hi == 0) vld = 1.0f - (float)mask[(bbase + lo) * kT + t];

      // Prefetch p 8 steps ahead: 32 lanes x 128B lines cover 16 rows x 256B.
      if (t + 8 < kT) {
        const char* pf = (const char*)(p + (size_t)(bbase + lo) * (kT * kD)
                                         + (size_t)(t + 8) * kD) + hi * 128;
        __builtin_prefetch(pf, 0, 1);
      }
    }

    // ---- final consume of u_{T-1} ----
    asm volatile("" ::: "memory");
    {
      const float* rowp = &uls[lo * kROW];
      float csum = 0.f;
      #pragma unroll
      for (int i = 0; i < 8; ++i) {
        csum += rowp[hi * 8 + i]      + rowp[16 + hi * 8 + i]
              + rowp[32 + hi * 8 + i] + rowp[48 + hi * 8 + i];
      }
      csum += __shfl_xor(csum, 16, 32);
      if (hi == 0) zacc += vld * __logf(csum);
    }

    if (hi == 0) out[bbase + lo] = zacc - s_ws[bbase + lo];
}

// ---------------------------------------------------------------------------
// Kernel 1: emission + transition scores (s1 + s2) per batch -> d_ws
// ---------------------------------------------------------------------------
__global__ __launch_bounds__(256)
void crf_score(const float* __restrict__ p,
               const int*   __restrict__ y,
               const int*   __restrict__ mask,
               const float* __restrict__ transition,
               float*       __restrict__ s_ws)
{
    __shared__ float wsum[8];
    const int b    = blockIdx.x;
    const int tid  = threadIdx.x;
    const int lane = tid & 31;
    const int wv   = tid >> 5;                  // wave 0..7, contiguous T chunks
    const int t0   = wv * (kT / 8);

    const size_t pb = (size_t)b * kT * kD;
    const int*   yb = y + pb;

    float s1 = 0.f, s2 = 0.f;
    int   prev_label = 0;
    float vprev = 0.f;

    if (wv > 0) {                               // label at chunk boundary t0-1
      const int tp = t0 - 1;
      const int yl = yb[(size_t)tp * kD + lane];
      const int yh = yb[(size_t)tp * kD + lane + 32];
      int lb = yl * lane + yh * (lane + 32);
      for (int off = 16; off > 0; off >>= 1) lb += __shfl_xor(lb, off, 32);
      prev_label = lb;
      vprev = 1.0f - (float)mask[b * kT + tp];
    }

    for (int t = t0; t < t0 + kT / 8; ++t) {
      const int   yl = yb[(size_t)t * kD + lane];
      const int   yh = yb[(size_t)t * kD + lane + 32];
      const float pl = p[pb + (size_t)t * kD + lane];
      const float ph = p[pb + (size_t)t * kD + lane + 32];
      const float vcur = 1.0f - (float)mask[b * kT + t];
      s1 += vcur * ((float)yl * pl + (float)yh * ph);   // per-lane partial
      int lb = yl * lane + yh * (lane + 32);            // one-hot -> label
      for (int off = 16; off > 0; off >>= 1) lb += __shfl_xor(lb, off, 32);
      if (lane == 0) s2 += vprev * vcur * transition[prev_label * kD + lb];
      prev_label = lb;
      vprev = vcur;
    }

    for (int off = 16; off > 0; off >>= 1) s1 += __shfl_xor(s1, off, 32);
    if (lane == 0) wsum[wv] = s1 + s2;
    __syncthreads();
    if (tid == 0) {
      float tot = 0.f;
      for (int w = 0; w < 8; ++w) tot += wsum[w];
      s_ws[b] = tot;
    }
}

// ---------------------------------------------------------------------------
extern "C" void kernel_launch(void* const* d_in, const int* in_sizes, int n_in,
                              void* d_out, int out_size, void* d_ws, size_t ws_size,
                              hipStream_t stream) {
    (void)in_sizes; (void)n_in; (void)out_size; (void)ws_size;
    const float* p          = (const float*)d_in[0];
    const int*   y          = (const int*)  d_in[1];
    const int*   mask       = (const int*)  d_in[2];
    const float* transition = (const float*)d_in[3];
    float* out  = (float*)d_out;
    float* s_ws = (float*)d_ws;          // 256 floats of scratch

    crf_score  <<<kB,        256, 0, stream>>>(p, y, mask, transition, s_ws);
    crf_forward<<<kB / kMB,   32, 0, stream>>>(p, mask, transition, s_ws, out);
}
